// BigramLanguageModel_37203006718446
// MI455X (gfx1250) — matrix operand order repaired and checked by reading, not compiled, not run on hardware
//
#include <hip/hip_runtime.h>
#include <math.h>

// ---------------------------------------------------------------------------
// Mini-GPT forward for gfx1250 (MI455X). All matmuls via v_wmma_f32_16x16x32_f16
// with LDS-staged f16 fragments (coalesced b128 global loads, b128 DS reads).
// GEMM is templated on its epilogue/layout flags so every instantiation has a
// straight-line staging path (no runtime flag branches in the K loop).
// Dims: L=6, H=6, HS=64, D=384, T=256, V=65, B=64, M=B*T=16384.
// Workspace requirement: (6*M*D + M*4*D + M) * 4 bytes ~= 240 MB in d_ws.
// ---------------------------------------------------------------------------

#define Lnum 6
#define Hn   6
#define HS   64
#define Dm   384
#define Tn   256
#define Vn   65
#define Bn   64
#define Mrows (Bn*Tn)        // 16384
#define NEG_INF (-__builtin_inff())

typedef _Float16 v16h __attribute__((ext_vector_type(16)));
typedef _Float16 v8h  __attribute__((ext_vector_type(8)));
typedef _Float16 v4h  __attribute__((ext_vector_type(4)));
typedef float    v8f  __attribute__((ext_vector_type(8)));

enum { GF_BIAS = 1, GF_RELU = 2, GF_RES = 4, GF_BHEADS = 8 };

__device__ __forceinline__ v8h cvt8(float4 a, float4 b) {
  v8h r;
  r[0] = (_Float16)a.x; r[1] = (_Float16)a.y;
  r[2] = (_Float16)a.z; r[3] = (_Float16)a.w;
  r[4] = (_Float16)b.x; r[5] = (_Float16)b.y;
  r[6] = (_Float16)b.z; r[7] = (_Float16)b.w;
  return r;
}
__device__ __forceinline__ v16h cat16(v8h lo, v8h hi) {
  return __builtin_shufflevector(lo, hi, 0, 1, 2, 3, 4, 5, 6, 7,
                                 8, 9, 10, 11, 12, 13, 14, 15);
}
__device__ __forceinline__ v8f wmma_f16(const v16h& a, const v16h& b, v8f c) {
  return __builtin_amdgcn_wmma_f32_16x16x32_f16(false, a, false, b, (short)0, c,
                                                false, false);
}
__device__ __forceinline__ void f4put(float* f, float4 a, int o) {
  f[o] = a.x; f[o + 1] = a.y; f[o + 2] = a.z; f[o + 3] = a.w;
}
// fragment = two 16-byte contiguous DS reads (A: K=g*16..+15)
__device__ __forceinline__ v16h afrag(const _Float16* sm, int row, int ld, int kb, int g) {
  return cat16(*reinterpret_cast<const v8h*>(&sm[row * ld + kb + g * 16]),
               *reinterpret_cast<const v8h*>(&sm[row * ld + kb + g * 16 + 8]));
}
// fragment = one 32-byte contiguous run (B^T: K=g*16..+15)
__device__ __forceinline__ v16h bfrag(const _Float16* sm, int row, int ld, int kb, int g) {
  return cat16(*reinterpret_cast<const v8h*>(&sm[row * ld + kb + g * 16]),
               *reinterpret_cast<const v8h*>(&sm[row * ld + kb + g * 16 + 8]));
}

// ---------------------------------------------------------------------------
// Embedding: x[m, d] = tok_emb[idx[m], d] + pos_emb[m % T, d]
// ---------------------------------------------------------------------------
__global__ void k_embed(const int* __restrict__ idx, const float* __restrict__ tok,
                        const float* __restrict__ pos, float* __restrict__ x) {
  int e = blockIdx.x * blockDim.x + threadIdx.x;
  if (e >= Mrows * Dm) return;
  int m = e / Dm, d = e - m * Dm;
  int t = m % Tn;
  x[e] = tok[idx[m] * Dm + d] + pos[t * Dm + d];
}

// ---------------------------------------------------------------------------
// LayerNorm: one wave per row of D=384 (12 elems / lane), wave32 reductions.
// ---------------------------------------------------------------------------
__global__ void k_layernorm(const float* __restrict__ x, const float* __restrict__ gam,
                            const float* __restrict__ bet, float* __restrict__ y) {
  int m = blockIdx.x * blockDim.y + threadIdx.y;
  if (m >= Mrows) return;
  int lane = threadIdx.x;
  float v[12];
  float s = 0.f;
#pragma unroll
  for (int i = 0; i < 12; ++i) { v[i] = x[(size_t)m * Dm + lane + 32 * i]; s += v[i]; }
#pragma unroll
  for (int off = 16; off > 0; off >>= 1) s += __shfl_xor(s, off, 32);
  float mean = s * (1.0f / Dm);
  float q = 0.f;
#pragma unroll
  for (int i = 0; i < 12; ++i) { float d = v[i] - mean; q += d * d; }
#pragma unroll
  for (int off = 16; off > 0; off >>= 1) q += __shfl_xor(q, off, 32);
  float rstd = rsqrtf(q * (1.0f / Dm) + 1e-5f);
#pragma unroll
  for (int i = 0; i < 12; ++i) {
    int c = lane + 32 * i;
    y[(size_t)m * Dm + c] = (v[i] - mean) * rstd * gam[c] + bet[c];
  }
}

// ---------------------------------------------------------------------------
// WMMA GEMM: C[M,N] = A[M,K] @ B[K,N] (+bias)(+relu)(+residual).
// 128 threads = 4 waves; block tile 32x64, K-step 64. Each wave owns a 16x32
// strip: one shared A fragment feeds two B fragments -> 4 WMMAs per K-step.
// LDS: A staged [m][k] f16, B staged transposed [n][k] f16; row stride 72
// halves keeps the 16-byte DS fragment reads bank-conflict-free.
// GF_BHEADS: B is the [H, K, 64] head-major weight; col c -> (c>>6, c&63).
// RAGGED (head, N=65): scalar guarded staging + guarded stores.
// Requires M % 32 == 0 and K % 64 == 0 (true for all uses here).
// ---------------------------------------------------------------------------
#define GBM 32
#define GBN 64
#define GBK 64
#define LDK 72

template <int FLAGS, bool RAGGED>
__global__ void __launch_bounds__(128)
k_gemm(const float* __restrict__ A, const float* __restrict__ Bm,
       const float* __restrict__ bias, const float* __restrict__ res,
       float* __restrict__ C, int M, int N, int K) {
  __shared__ __align__(16) _Float16 sA[GBM * LDK];
  __shared__ __align__(16) _Float16 sB[GBN * LDK];
  const int lane = threadIdx.x, ty = threadIdx.y;
  const int t = ty * 32 + lane;
  const int nl = lane & 15, g = lane >> 4;
  const int wm = ty >> 1, wn = ty & 1;
  const int m0 = blockIdx.y * GBM, n0 = blockIdx.x * GBN;

  const int ar = t >> 2, ac = (t & 3) * 16;   // A: row 0..31, k-chunk {0,16,32,48}
  const int bk = t >> 1, bc = (t & 1) * 32;   // B: k 0..63,  n-chunk {0,32}

  v8f acc0 = {}, acc1 = {};
  for (int k0 = 0; k0 < K; k0 += GBK) {
    { // stage A: 16 consecutive floats -> two v8h stores
      const float4* p =
          reinterpret_cast<const float4*>(A + (size_t)(m0 + ar) * K + k0 + ac);
      float4 q0 = p[0], q1 = p[1], q2 = p[2], q3 = p[3];
      *reinterpret_cast<v8h*>(&sA[ar * LDK + ac])     = cvt8(q0, q1);
      *reinterpret_cast<v8h*>(&sA[ar * LDK + ac + 8]) = cvt8(q2, q3);
      if (k0 + GBK < K)   // prefetch next A panel (global_prefetch_b8)
        __builtin_prefetch(A + (size_t)(m0 + ar) * K + k0 + GBK + ac, 0, 0);
    }
    { // stage B transposed into sB[n][k]: 32 columns / thread
      const int kk = k0 + bk;
      const int c0 = n0 + bc;
      float f[32];
      if constexpr (RAGGED) {
#pragma unroll
        for (int i = 0; i < 32; ++i) {
          int c = c0 + i;
          f[i] = (c < N) ? Bm[(size_t)kk * N + c] : 0.f;
        }
      } else if constexpr ((FLAGS & GF_BHEADS) != 0) {
        const float4* p = reinterpret_cast<const float4*>(
            Bm + (((size_t)(c0 >> 6)) * K + kk) * 64 + (c0 & 63));
#pragma unroll
        for (int j = 0; j < 8; ++j) f4put(f, p[j], 4 * j);
      } else {
        const float4* p =
            reinterpret_cast<const float4*>(Bm + (size_t)kk * N + c0);
#pragma unroll
        for (int j = 0; j < 8; ++j) f4put(f, p[j], 4 * j);
      }
#pragma unroll
      for (int i = 0; i < 32; ++i) sB[(bc + i) * LDK + bk] = (_Float16)f[i];
    }
    __syncthreads();
#pragma unroll
    for (int s = 0; s < 2; ++s) {
      const int kb = s * 32;
      v16h av  = afrag(sA, wm * 16 + nl, LDK, kb, g);
      v16h bv0 = bfrag(sB, wn * 32 + nl, LDK, kb, g);
      v16h bv1 = bfrag(sB, wn * 32 + 16 + nl, LDK, kb, g);
      acc0 = wmma_f16(av, bv0, acc0);
      acc1 = wmma_f16(av, bv1, acc1);
    }
    __syncthreads();
  }
  const int col0 = n0 + wn * 32 + nl;
  const int col1 = col0 + 16;
#pragma unroll
  for (int r = 0; r < 8; ++r) {
    const int m = m0 + wm * 16 + g * 8 + r;
    float v0 = acc0[r], v1 = acc1[r];
    if constexpr ((FLAGS & GF_BIAS) != 0) {
      if (!RAGGED || col0 < N) v0 += bias[col0];
      if (!RAGGED || col1 < N) v1 += bias[col1];
    }
    if constexpr ((FLAGS & GF_RELU) != 0) { v0 = fmaxf(v0, 0.f); v1 = fmaxf(v1, 0.f); }
    if constexpr ((FLAGS & GF_RES) != 0) {
      v0 += res[(size_t)m * N + col0];
      v1 += res[(size_t)m * N + col1];
    }
    if (!RAGGED || col0 < N) C[(size_t)m * N + col0] = v0;
    if (!RAGGED || col1 < N) C[(size_t)m * N + col1] = v1;
  }
}

// ---------------------------------------------------------------------------
// Attention scores: S[bh, ti, tj] = (q . k) * scale (causal mask in softmax).
// Block: 4 waves cover one ti tile x four tj tiles; q tile (16x64) and k tile
// (64x64) staged once. k's [B,T,H,HS] layout is already the transposed-B
// layout sK[n][kk], so staging needs no transpose.
// ---------------------------------------------------------------------------
__global__ void __launch_bounds__(128)
k_attn_scores(const float* __restrict__ q, const float* __restrict__ kq,
              float* __restrict__ S, float scale) {
  __shared__ __align__(16) _Float16 sQ[16 * LDK];
  __shared__ __align__(16) _Float16 sK[64 * LDK];
  const int bh = blockIdx.z;
  const int bb = bh / Hn, hd = bh % Hn;
  const int ti0 = blockIdx.y * 16;
  const int tj00 = blockIdx.x * 64;
  if (tj00 > ti0 + 15) return;          // whole block above diagonal (uniform)
  const int lane = threadIdx.x, ty = threadIdx.y;
  const int t = ty * 32 + lane;
  const int nl = lane & 15, g = lane >> 4;

  { // stage Q: 16 rows x 64; 8 floats / thread
    int r = t >> 3, c = (t & 7) * 8;
    const float4* p = reinterpret_cast<const float4*>(
        q + (((size_t)(bb * Tn + ti0 + r)) * Hn + hd) * HS + c);
    float4 q0 = p[0], q1 = p[1];
    *reinterpret_cast<v8h*>(&sQ[r * LDK + c]) = cvt8(q0, q1);
  }
  { // stage K: 64 rows x 64; 32 floats / thread
    int r = t >> 1, c = (t & 1) * 32;
    const float4* p = reinterpret_cast<const float4*>(
        kq + (((size_t)(bb * Tn + tj00 + r)) * Hn + hd) * HS + c);
#pragma unroll
    for (int j = 0; j < 4; ++j) {
      float4 q0 = p[2 * j], q1 = p[2 * j + 1];
      *reinterpret_cast<v8h*>(&sK[r * LDK + c + 8 * j]) = cvt8(q0, q1);
    }
  }
  __syncthreads();

  v8f acc = {};
#pragma unroll
  for (int k0 = 0; k0 < HS; k0 += 32) {
    v16h av = afrag(sQ, nl, LDK, k0, g);
    v16h bv = bfrag(sK, ty * 16 + nl, LDK, k0, g);
    acc = wmma_f16(av, bv, acc);
  }
  const int tj0 = tj00 + ty * 16;
#pragma unroll
  for (int r = 0; r < 8; ++r) {
    int ti = ti0 + g * 8 + r;
    S[((size_t)bh * Tn + ti) * Tn + tj0 + nl] = acc[r] * scale;
  }
}

// ---------------------------------------------------------------------------
// Causal row softmax over T=256 (8 elems/lane); writes 0 at masked positions
// so P@V can read anywhere and truncate its K loop at the diagonal.
// ---------------------------------------------------------------------------
__global__ void k_softmax(float* __restrict__ S) {
  int ridx = blockIdx.x * blockDim.y + threadIdx.y;   // over B*H*T rows
  if (ridx >= Bn * Hn * Tn) return;
  int ti = ridx % Tn;
  int lane = threadIdx.x;
  float vals[8];
  float mx = NEG_INF;
#pragma unroll
  for (int s = 0; s < 8; ++s) {
    int j = lane + 32 * s;
    float vv = (j <= ti) ? S[(size_t)ridx * Tn + j] : NEG_INF;
    vals[s] = vv;
    mx = fmaxf(mx, vv);
  }
#pragma unroll
  for (int off = 16; off > 0; off >>= 1) mx = fmaxf(mx, __shfl_xor(mx, off, 32));
  float sum = 0.f;
#pragma unroll
  for (int s = 0; s < 8; ++s) {
    int j = lane + 32 * s;
    float e = (j <= ti) ? expf(vals[s] - mx) : 0.f;
    vals[s] = e;
    sum += e;
  }
#pragma unroll
  for (int off = 16; off > 0; off >>= 1) sum += __shfl_xor(sum, off, 32);
  float inv = 1.0f / sum;
#pragma unroll
  for (int s = 0; s < 8; ++s) S[(size_t)ridx * Tn + lane + 32 * s] = vals[s] * inv;
}

// ---------------------------------------------------------------------------
// O = P @ V, K loop truncated at the causal diagonal (P zero beyond it).
// Block: 4 waves cover one ti tile x all of HS (4 n-tiles). P rows are
// contiguous in S; V is transposed into LDS during staging.
// Output layout [B, T, H, HS] == concat-heads.
// ---------------------------------------------------------------------------
#define LDP 40

__global__ void __launch_bounds__(128)
k_attn_pv(const float* __restrict__ S, const float* __restrict__ vm,
          float* __restrict__ o) {
  __shared__ __align__(16) _Float16 sP[16 * LDP];
  __shared__ __align__(16) _Float16 sV[64 * LDP];
  const int bh = blockIdx.z;
  const int bb = bh / Hn, hd = bh % Hn;
  const int ti0 = blockIdx.y * 16;
  const int lane = threadIdx.x, ty = threadIdx.y;
  const int t = ty * 32 + lane;
  const int nl = lane & 15, g = lane >> 4;

  v8f acc = {};
  const int kend = ti0 + 16;            // uniform over block
  for (int k0 = 0; k0 < kend; k0 += 32) {
    { // stage P: 16 rows x 32; 4 floats / thread
      int r = t >> 3, c = (t & 7) * 4;
      float4 f = *reinterpret_cast<const float4*>(
          S + ((size_t)bh * Tn + ti0 + r) * Tn + k0 + c);
      v4h h;
      h[0] = (_Float16)f.x; h[1] = (_Float16)f.y;
      h[2] = (_Float16)f.z; h[3] = (_Float16)f.w;
      *reinterpret_cast<v4h*>(&sP[r * LDP + c]) = h;
    }
    { // stage V transposed: sV[n][kk]; 16 floats / thread
      int kk = t >> 2, c = (t & 3) * 16;
      const float4* p = reinterpret_cast<const float4*>(
          vm + (((size_t)(bb * Tn + k0 + kk)) * Hn + hd) * HS + c);
      float f[16];
      float4 q0 = p[0], q1 = p[1], q2 = p[2], q3 = p[3];
      f4put(f, q0, 0); f4put(f, q1, 4); f4put(f, q2, 8); f4put(f, q3, 12);
#pragma unroll
      for (int i = 0; i < 16; ++i) sV[(c + i) * LDP + kk] = (_Float16)f[i];
    }
    __syncthreads();
    v16h av = afrag(sP, nl, LDP, 0, g);
    v16h bv = bfrag(sV, ty * 16 + nl, LDP, 0, g);
    acc = wmma_f16(av, bv, acc);
    __syncthreads();
  }
#pragma unroll
  for (int r = 0; r < 8; ++r) {
    int ti = ti0 + g * 8 + r;
    o[(((size_t)(bb * Tn + ti)) * Hn + hd) * HS + ty * 16 + nl] = acc[r];
  }
}

// ---------------------------------------------------------------------------
// Per-token NLL: one wave per token (V=65, 3 elems/lane), pre-scaled by 1/M.
// ---------------------------------------------------------------------------
__global__ void k_nll(const float* __restrict__ logits, const int* __restrict__ tgt,
                      float* __restrict__ nll) {
  int m = blockIdx.x * blockDim.y + threadIdx.y;
  if (m >= Mrows) return;
  int lane = threadIdx.x;
  float lv[3];
  float mx = NEG_INF;
#pragma unroll
  for (int s = 0; s < 3; ++s) {
    int j = lane + 32 * s;
    lv[s] = (j < Vn) ? logits[(size_t)m * Vn + j] : NEG_INF;
    mx = fmaxf(mx, lv[s]);
  }
#pragma unroll
  for (int off = 16; off > 0; off >>= 1) mx = fmaxf(mx, __shfl_xor(mx, off, 32));
  float sum = 0.f;
#pragma unroll
  for (int s = 0; s < 3; ++s) {
    int j = lane + 32 * s;
    sum += (j < Vn) ? expf(lv[s] - mx) : 0.f;
  }
#pragma unroll
  for (int off = 16; off > 0; off >>= 1) sum += __shfl_xor(sum, off, 32);
  if (lane == 0) {
    float lse = mx + logf(sum);
    nll[m] = (lse - logits[(size_t)m * Vn + tgt[m]]) * (1.0f / Mrows);
  }
}

// Deterministic final reduction (fixed order -> replay-stable).
__global__ void k_loss_reduce(const float* __restrict__ nll, float* __restrict__ out) {
  __shared__ float sh[256];
  int tid = threadIdx.x;
  float s = 0.f;
  for (int i = tid; i < Mrows; i += 256) s += nll[i];
  sh[tid] = s;
  __syncthreads();
  for (int off = 128; off > 0; off >>= 1) {
    if (tid < off) sh[tid] += sh[tid + off];
    __syncthreads();
  }
  if (tid == 0) *out = sh[0];
}

// ---------------------------------------------------------------------------
extern "C" void kernel_launch(void* const* d_in, const int* in_sizes, int n_in,
                              void* d_out, int out_size, void* d_ws, size_t ws_size,
                              hipStream_t stream) {
  const int*   idx   = (const int*)d_in[0];
  const int*   tgt   = (const int*)d_in[1];
  const float* tok   = (const float*)d_in[2];
  const float* pos   = (const float*)d_in[3];
  const float* Wq    = (const float*)d_in[4];
  const float* Wk    = (const float*)d_in[5];
  const float* Wv    = (const float*)d_in[6];
  const float* Wproj = (const float*)d_in[7];
  const float* bproj = (const float*)d_in[8];
  const float* W1    = (const float*)d_in[9];
  const float* b1    = (const float*)d_in[10];
  const float* W2    = (const float*)d_in[11];
  const float* b2    = (const float*)d_in[12];
  const float* ln1g  = (const float*)d_in[13];
  const float* ln1b  = (const float*)d_in[14];
  const float* lnfg  = (const float*)d_in[15];
  const float* lnfb  = (const float*)d_in[16];
  const float* Whead = (const float*)d_in[17];
  const float* bhead = (const float*)d_in[18];

  float* logits = (float*)d_out;                        // [M, V]
  float* lossp  = logits + (size_t)Mrows * Vn;          // scalar

  // Workspace carve-up (floats).
  float* ws = (float*)d_ws;
  const size_t MD = (size_t)Mrows * Dm;
  float* x   = ws;               // [M, D] residual stream
  float* y   = x + MD;           // [M, D] LN output
  float* q   = y + MD;           // [B, T, H, HS]
  float* k   = q + MD;
  float* v   = k + MD;
  float* o   = v + MD;           // attention output, concat-head layout
  float* big = o + MD;           // M*4D floats: attn scores [B*H,T,T] / FFN hidden
  float* nll = big + (size_t)Mrows * 4 * Dm;  // [M]

  const dim3 blk(32, 4);                       // 4 wave32 per block
  const float scale = 1.0f / sqrtf((float)Dm); // reference scales by D**-0.5

  { // embedding
    int total = Mrows * Dm;
    k_embed<<<(total + 255) / 256, 256, 0, stream>>>(idx, tok, pos, x);
  }

  const dim3 gD(Dm / GBN, Mrows / GBM);                  // N=384 GEMMs
  const dim3 gF(4 * Dm / GBN, Mrows / GBM);              // N=1536 GEMM
  const dim3 gH((Vn + GBN - 1) / GBN, Mrows / GBM);      // N=65 head GEMM
  const dim3 gS(Tn / 64, Tn / 16, Bn * Hn);              // score tiles
  const dim3 gPV(1, Tn / 16, Bn * Hn);                   // P@V tiles

  for (int l = 0; l < Lnum; ++l) {
    const float* g1 = ln1g + (size_t)l * Dm;
    const float* o1 = ln1b + (size_t)l * Dm;

    k_layernorm<<<Mrows / 4, blk, 0, stream>>>(x, g1, o1, y);

    const size_t wqk = (size_t)l * Hn * Dm * HS;
    k_gemm<GF_BHEADS, false><<<gD, blk, 0, stream>>>(
        y, Wq + wqk, nullptr, nullptr, q, Mrows, Dm, Dm);
    k_gemm<GF_BHEADS, false><<<gD, blk, 0, stream>>>(
        y, Wk + wqk, nullptr, nullptr, k, Mrows, Dm, Dm);
    k_gemm<GF_BHEADS, false><<<gD, blk, 0, stream>>>(
        y, Wv + wqk, nullptr, nullptr, v, Mrows, Dm, Dm);

    k_attn_scores<<<gS, blk, 0, stream>>>(q, k, big, scale);
    k_softmax<<<(Bn * Hn * Tn) / 4, blk, 0, stream>>>(big);
    k_attn_pv<<<gPV, blk, 0, stream>>>(big, v, o);

    k_gemm<GF_BIAS | GF_RES, false><<<gD, blk, 0, stream>>>(
        o, Wproj + (size_t)l * Dm * Dm, bproj + (size_t)l * Dm, x, x,
        Mrows, Dm, Dm);

    // faithful to reference: reuse ln1 params before FFN
    k_layernorm<<<Mrows / 4, blk, 0, stream>>>(x, g1, o1, y);

    k_gemm<GF_BIAS | GF_RELU, false><<<gF, blk, 0, stream>>>(
        y, W1 + (size_t)l * Dm * 4 * Dm, b1 + (size_t)l * 4 * Dm, nullptr, big,
        Mrows, 4 * Dm, Dm);
    k_gemm<GF_BIAS | GF_RES, false><<<gD, blk, 0, stream>>>(
        big, W2 + (size_t)l * 4 * Dm * Dm, b2 + (size_t)l * Dm, x, x,
        Mrows, Dm, 4 * Dm);
  }

  k_layernorm<<<Mrows / 4, blk, 0, stream>>>(x, lnfg, lnfb, y);

  k_gemm<GF_BIAS, true><<<gH, blk, 0, stream>>>(
      y, Whead, bhead, nullptr, logits, Mrows, Vn, Dm);

  k_nll<<<Mrows / 4, blk, 0, stream>>>(logits, tgt, nll);
  k_loss_reduce<<<1, 256, 0, stream>>>(nll, lossp);
}
